// Self_Attn_12661563588950
// MI455X (gfx1250) — matrix operand order repaired and checked by reading, not compiled, not run on hardware
//
#include <hip/hip_runtime.h>
#include <math.h>

// ---------------------------------------------------------------------------
// CDNA5 (gfx1250, wave32) implementation of the Self_Attn reference.
//
// ~145 GFLOP total vs ~0.5 GB HBM traffic -> roughly balanced on MI455X.
// All 1x1 convs and attention matmuls run through one tiled bf16 WMMA GEMM
// (v_wmma_f32_16x16x32_bf16, fp32 accum), 128x128 block tiles, 8 waves/block,
// 32x64 per wave, double-buffered LDS fed by GLOBAL_LOAD_ASYNC_TO_LDS_B128
// (ASYNCcnt-tracked) when the toolchain exposes the builtin.
// ---------------------------------------------------------------------------

typedef __attribute__((ext_vector_type(16))) __bf16 v16bf;
typedef __attribute__((ext_vector_type(8)))  float  v8f;
typedef int v4i __attribute__((vector_size(16)));   // matches builtin param type

#define BM 128
#define BN 128
#define BK 32

// Problem constants
#define NB    64      // B*T
#define CIN   2048
#define CRR   256
#define HW    196     // 14*14
#define NPIX  12544   // NB*HW
#define SS    3136    // T*H*W
#define BB    4

#define AS1 __attribute__((address_space(1)))
#define AS3 __attribute__((address_space(3)))

#if __has_builtin(__builtin_amdgcn_global_load_async_to_lds_b128)
#define ASYNC_LDS 1
#endif

__device__ __forceinline__ void wait_async0() {
#if defined(ASYNC_LDS)
#if __has_builtin(__builtin_amdgcn_s_wait_asynccnt)
  __builtin_amdgcn_s_wait_asynccnt(0);
#else
  asm volatile("s_wait_asynccnt 0x0" ::: "memory");
#endif
#endif
}

union FragBF { uint4 u[2]; v16bf v; };

// ---------------------------------------------------------------------------
// Generic GEMM: C[M,N] = A[M,K](bf16,row-major) * Bt[N,K](bf16,row-major)^T
//               (+ bias[N] if bias != nullptr), batched via blockIdx.z.
// Staging uses index CLAMPING instead of predication: a clamped (duplicate)
// A-row only feeds C-rows >= M, which are never stored; same for B-cols.
// This keeps EXEC all-ones (required for async-to-LDS and WMMA).
// ---------------------------------------------------------------------------
template <typename OutT>
__global__ __launch_bounds__(256) void gemm_bt_kernel(
    const __bf16* __restrict__ A, const __bf16* __restrict__ Bt,
    OutT* __restrict__ C, const float* __restrict__ bias,
    int M, int N, int K,
    long long strideA, long long strideB, long long strideC)
{
  __shared__ __align__(16) __bf16 sA[2][BM][BK];   // 2 x 8 KB
  __shared__ __align__(16) __bf16 sB[2][BN][BK];   // 2 x 8 KB

  const int bz = blockIdx.z;
  A  += (long long)bz * strideA;
  Bt += (long long)bz * strideB;
  C  += (long long)bz * strideC;

  const int tid  = threadIdx.x;
  const int lane = tid & 31;
  const int wave = tid >> 5;            // 0..7
  const int wm   = wave & 3;            // 4 waves along M (32 rows each)
  const int wn   = wave >> 2;           // 2 waves along N (64 cols each)
  const int hi   = lane >> 4;           // 0: lanes 0-15, 1: lanes 16-31
  const int l15  = lane & 15;

  const int blockM = blockIdx.y * BM;
  const int blockN = blockIdx.x * BN;

  v8f acc[2][4];
#pragma unroll
  for (int i = 0; i < 2; ++i)
#pragma unroll
    for (int j = 0; j < 4; ++j)
#pragma unroll
      for (int r = 0; r < 8; ++r) acc[i][j][r] = 0.0f;

  // Stage one 128x32 A tile + 128x32 Bt tile into LDS buffer `buf`.
  auto stage = [&](int k0, int buf) {
#pragma unroll
    for (int c = tid; c < 512; c += 256) {
      const int row = c >> 2, seg = c & 3;
      int grow = blockM + row; if (grow >= M) grow = M - 1;   // clamp, no branch
      const __bf16* gp = A + (long long)grow * K + k0 + seg * 8;
#if defined(ASYNC_LDS)
      __builtin_amdgcn_global_load_async_to_lds_b128(
          (AS1 v4i*)gp, (AS3 v4i*)&sA[buf][row][seg * 8], 0, 0);
#else
      *(uint4*)&sA[buf][row][seg * 8] = *(const uint4*)gp;
#endif
      if (seg == 0) __builtin_prefetch(gp + 2 * BK, 0, 3);    // global_prefetch_b8
    }
#pragma unroll
    for (int c = tid; c < 512; c += 256) {
      const int row = c >> 2, seg = c & 3;
      int gn = blockN + row; if (gn >= N) gn = N - 1;         // clamp, no branch
      const __bf16* gp = Bt + (long long)gn * K + k0 + seg * 8;
#if defined(ASYNC_LDS)
      __builtin_amdgcn_global_load_async_to_lds_b128(
          (AS1 v4i*)gp, (AS3 v4i*)&sB[buf][row][seg * 8], 0, 0);
#else
      *(uint4*)&sB[buf][row][seg * 8] = *(const uint4*)gp;
#endif
    }
  };

  const int nT = K / BK;                 // K is always a multiple of 32 here
  stage(0, 0);

  for (int t = 0; t < nT; ++t) {
    wait_async0();                       // my async stores to LDS have landed
    __syncthreads();                     // everyone's landed; prev reads done
    if (t + 1 < nT) stage((t + 1) * BK, (t + 1) & 1);   // overlap next tile

    const int buf = t & 1;
    // A fragments (16x32 bf16): lane<16 holds K{0..7,16..23}, lane>=16 holds
    // K{8..15,24..31} of row M=l15 (ISA 7.12.2 16-bit A layout).
    FragBF fa[2];
#pragma unroll
    for (int i = 0; i < 2; ++i) {
      const __bf16* p = &sA[buf][wm * 32 + i * 16 + l15][hi * 8];
      fa[i].u[0] = *(const uint4*)p;
      fa[i].u[1] = *(const uint4*)(p + 16);
    }
    // B fragments (32x16 bf16): N=l15; lane<16 holds K{0..7,16..23}, lane>=16
    // holds K{8..15,24..31} (same K order as A).
    FragBF fb[4];
#pragma unroll
    for (int j = 0; j < 4; ++j) {
      const __bf16* p = &sB[buf][wn * 64 + j * 16 + l15][hi * 8];
      fb[j].u[0] = *(const uint4*)p;
      fb[j].u[1] = *(const uint4*)(p + 16);
    }

#pragma unroll
    for (int i = 0; i < 2; ++i)
#pragma unroll
      for (int j = 0; j < 4; ++j)
        acc[i][j] = __builtin_amdgcn_wmma_f32_16x16x32_bf16(
            false, fa[i].v, false, fb[j].v, (short)0, acc[i][j], false, false);
  }

  // Epilogue: C/D layout (ISA 7.12.2): VGPR r holds M=r+8*hi, N=l15.
#pragma unroll
  for (int i = 0; i < 2; ++i) {
#pragma unroll
    for (int j = 0; j < 4; ++j) {
      const int col = blockN + wn * 64 + j * 16 + l15;
      if (col >= N) continue;
      const float bv = bias ? bias[col] : 0.0f;
#pragma unroll
      for (int r = 0; r < 8; ++r) {
        const int row = blockM + wm * 32 + i * 16 + r + hi * 8;
        if (row < M) {
          const float val = acc[i][j][r] + bv;
          C[(long long)row * N + col] = (OutT)val;
        }
      }
    }
  }
}

// ---------------------------------------------------------------------------
// Elementwise / data-movement kernels
// ---------------------------------------------------------------------------

// NCHW f32 -> pixel-major bf16 [N*HW, C]
__global__ void nchw_to_pm(const float* __restrict__ in, __bf16* __restrict__ out,
                           long long total) {
  for (long long i = blockIdx.x * (long long)blockDim.x + threadIdx.x; i < total;
       i += (long long)gridDim.x * blockDim.x) {
    const long long c  = i % CIN;
    const long long p  = i / CIN;       // n*HW + hw
    const long long hw = p % HW;
    const long long n  = p / HW;
    out[i] = (__bf16)in[(n * CIN + c) * HW + hw];
  }
}

__global__ void cvt_f32_bf16(const float* __restrict__ in, __bf16* __restrict__ out,
                             long long total) {
  for (long long i = blockIdx.x * (long long)blockDim.x + threadIdx.x; i < total;
       i += (long long)gridDim.x * blockDim.x)
    out[i] = (__bf16)in[i];
}

// Reproduce the reference's k scramble: flat(h,t,w,cr) reinterpreted as (c,s)
// with S=3136. We emit the TRANSPOSED view kT[b][s][c] for the GEMM Bt input.
__global__ void k_scramble(const __bf16* __restrict__ kraw_pm,
                           __bf16* __restrict__ kT) {
  const long long total = (long long)BB * SS * CRR;
  for (long long i = blockIdx.x * (long long)blockDim.x + threadIdx.x; i < total;
       i += (long long)gridDim.x * blockDim.x) {
    const long long c  = i % CRR;
    const long long bs = i / CRR;
    const long long s  = bs % SS;
    const long long b  = bs / SS;
    const long long pos = c * SS + s;        // flat index into [H,T,W,CR]
    const long long h   = pos / (16LL * 14 * 256);   // 57344
    const long long r1  = pos % (16LL * 14 * 256);
    const long long t   = r1 / (14LL * 256);         // 3584
    const long long r2  = r1 % (14LL * 256);
    const long long w   = r2 / 256;
    const long long cr  = r2 % 256;
    const long long m   = (b * 16 + t) * HW + h * 14 + w;
    kT[i] = kraw_pm[m * CRR + cr];
  }
}

// v [B*S, CR] -> vT [b][CR, S]
__global__ void transpose_sc(const __bf16* __restrict__ v, __bf16* __restrict__ vT) {
  const long long total = (long long)BB * SS * CRR;
  for (long long i = blockIdx.x * (long long)blockDim.x + threadIdx.x; i < total;
       i += (long long)gridDim.x * blockDim.x) {
    const long long s  = i % SS;
    const long long bc = i / SS;
    const long long cr = bc % CRR;
    const long long b  = bc / CRR;
    vT[i] = v[(b * SS + s) * CRR + cr];
  }
}

// Row softmax: fp32 logits row -> bf16 attention row. One 256-thread block/row.
__global__ __launch_bounds__(256) void softmax_rows(const float* __restrict__ logits,
                                                    __bf16* __restrict__ att,
                                                    int ncols) {
  const long long row = blockIdx.x;
  const float* in = logits + row * (long long)ncols;
  __bf16* out = att + row * (long long)ncols;
  __shared__ float red[8];
  const int tid = threadIdx.x, lane = tid & 31, w = tid >> 5;

  float m = -3.4e38f;
  for (int c = tid; c < ncols; c += 256) m = fmaxf(m, in[c]);
#pragma unroll
  for (int off = 16; off > 0; off >>= 1) m = fmaxf(m, __shfl_xor(m, off, 32));
  if (lane == 0) red[w] = m;
  __syncthreads();
  float mall = red[0];
#pragma unroll
  for (int i = 1; i < 8; ++i) mall = fmaxf(mall, red[i]);
  __syncthreads();

  float s = 0.f;
  for (int c = tid; c < ncols; c += 256) s += __expf(in[c] - mall);
#pragma unroll
  for (int off = 16; off > 0; off >>= 1) s += __shfl_xor(s, off, 32);
  if (lane == 0) red[w] = s;
  __syncthreads();
  float sall = 0.f;
#pragma unroll
  for (int i = 0; i < 8; ++i) sall += red[i];
  const float inv = 1.0f / sall;

  for (int c = tid; c < ncols; c += 256)
    out[c] = (__bf16)(__expf(in[c] - mall) * inv);
}

// out(NCHW,f32) = gamma * final_pm + x(NCHW,f32)
__global__ void epilogue_k(const float* __restrict__ final_pm,
                           const float* __restrict__ x,
                           const float* __restrict__ gamma,
                           float* __restrict__ out, long long total) {
  const float g = gamma[0];
  for (long long i = blockIdx.x * (long long)blockDim.x + threadIdx.x; i < total;
       i += (long long)gridDim.x * blockDim.x) {
    const long long hw = i % HW;
    const long long nc = i / HW;
    const long long c  = nc % CIN;
    const long long n  = nc / CIN;
    out[i] = g * final_pm[(n * HW + hw) * CIN + c] + x[i];
  }
}

// ---------------------------------------------------------------------------
// Host-side orchestration
// ---------------------------------------------------------------------------
static inline size_t align_up(size_t x) { return (x + 255) & ~(size_t)255; }

extern "C" void kernel_launch(void* const* d_in, const int* in_sizes, int n_in,
                              void* d_out, int out_size, void* d_ws, size_t ws_size,
                              hipStream_t stream) {
  (void)in_sizes; (void)n_in; (void)out_size; (void)ws_size;

  const float* x       = (const float*)d_in[0];
  const float* domainX = (const float*)d_in[1];
  const float* Wq      = (const float*)d_in[2];
  const float* bq      = (const float*)d_in[3];
  const float* Wk      = (const float*)d_in[4];
  const float* bk      = (const float*)d_in[5];
  const float* Wv      = (const float*)d_in[6];
  const float* bv      = (const float*)d_in[7];
  const float* Wval    = (const float*)d_in[8];
  const float* bval    = (const float*)d_in[9];
  const float* b_l1d   = (const float*)d_in[10];
  const float* b_l1u   = (const float*)d_in[11];
  const float* W_l2d   = (const float*)d_in[12];
  const float* b_l2d   = (const float*)d_in[13];
  const float* W_l2u   = (const float*)d_in[14];
  const float* b_l2u   = (const float*)d_in[15];
  const float* gamma   = (const float*)d_in[16];
  float* out = (float*)d_out;

  // Workspace carving
  char* ws = (char*)d_ws;
  size_t off = 0;
  auto alloc = [&](size_t bytes) -> void* {
    void* p = ws + off; off = align_up(off + bytes); return p;
  };
  const size_t szBigBF  = (size_t)NPIX * CIN * sizeof(__bf16);   // 51.4 MB
  const size_t szSmlBF  = (size_t)NPIX * CRR * sizeof(__bf16);   // 6.4 MB
  const size_t szW      = (size_t)CIN * CRR * sizeof(__bf16);    // 1 MB
  const size_t szLogit  = (size_t)BB * SS * SS * sizeof(float);  // 157 MB
  const size_t szFinal  = (size_t)NPIX * CIN * sizeof(float);    // 102.8 MB
  const size_t szAtt    = (size_t)BB * SS * SS * sizeof(__bf16); // 78.6 MB

  __bf16* x_pm  = (__bf16*)alloc(szBigBF);
  __bf16* big1  = (__bf16*)alloc(szBigBF);   // dX_pm -> d2 -> d4
  __bf16* w_q   = (__bf16*)alloc(szW);
  __bf16* w_k   = (__bf16*)alloc(szW);
  __bf16* w_v   = (__bf16*)alloc(szW);
  __bf16* w_l2d = (__bf16*)alloc(szW);
  __bf16* w_val = (__bf16*)alloc(szW);
  __bf16* w_l2u = (__bf16*)alloc(szW);
  __bf16* sml0  = (__bf16*)alloc(szSmlBF);   // d1 -> q
  __bf16* sml1  = (__bf16*)alloc(szSmlBF);   // d3 -> kraw -> vT
  __bf16* sml2  = (__bf16*)alloc(szSmlBF);   // kT -> attout
  __bf16* sml3  = (__bf16*)alloc(szSmlBF);   // v
  float*  bigL  = (float*)alloc(szLogit > szFinal ? szLogit : szFinal); // logits -> final
  __bf16* att   = (__bf16*)alloc(szAtt);

  const long long totBig = (long long)NPIX * CIN;

  // 1) canonicalize activations to pixel-major bf16
  nchw_to_pm<<<4096, 256, 0, stream>>>(x,       x_pm, totBig);
  nchw_to_pm<<<4096, 256, 0, stream>>>(domainX, big1, totBig);
  // 2) weights -> bf16 (already in [N,K] row-major = Bt layout)
  const long long wElems = (long long)CIN * CRR;
  cvt_f32_bf16<<<512, 256, 0, stream>>>(Wq,    w_q,   wElems);
  cvt_f32_bf16<<<512, 256, 0, stream>>>(Wk,    w_k,   wElems);
  cvt_f32_bf16<<<512, 256, 0, stream>>>(Wv,    w_v,   wElems);
  cvt_f32_bf16<<<512, 256, 0, stream>>>(W_l2d, w_l2d, wElems);
  cvt_f32_bf16<<<512, 256, 0, stream>>>(Wval,  w_val, wElems);
  cvt_f32_bf16<<<512, 256, 0, stream>>>(W_l2u, w_l2u, wElems);

  auto gemm_bf16 = [&](const __bf16* A, const __bf16* Bt, __bf16* C,
                       const float* bias, int M, int N, int K, int batch,
                       long long sA, long long sB, long long sC) {
    dim3 g((N + BN - 1) / BN, (M + BM - 1) / BM, batch);
    gemm_bt_kernel<__bf16><<<g, 256, 0, stream>>>(A, Bt, C, bias, M, N, K, sA, sB, sC);
  };
  auto gemm_f32 = [&](const __bf16* A, const __bf16* Bt, float* C,
                      const float* bias, int M, int N, int K, int batch,
                      long long sA, long long sB, long long sC) {
    dim3 g((N + BN - 1) / BN, (M + BM - 1) / BM, batch);
    gemm_bt_kernel<float><<<g, 256, 0, stream>>>(A, Bt, C, bias, M, N, K, sA, sB, sC);
  };

  // 3) domain pathway (weight sharing per reference)
  gemm_bf16(big1, w_q,   sml0, b_l1d, NPIX, CRR, CIN, 1, 0, 0, 0);   // d1
  gemm_bf16(sml0, w_val, big1, b_l1u, NPIX, CIN, CRR, 1, 0, 0, 0);   // d2
  gemm_bf16(big1, w_l2d, sml1, b_l2d, NPIX, CRR, CIN, 1, 0, 0, 0);   // d3
  gemm_bf16(sml1, w_l2u, big1, b_l2u, NPIX, CIN, CRR, 1, 0, 0, 0);   // d4

  // 4) q,k,v projections
  gemm_bf16(big1, w_q, sml0, bq, NPIX, CRR, CIN, 1, 0, 0, 0);        // q
  gemm_bf16(x_pm, w_k, sml1, bk, NPIX, CRR, CIN, 1, 0, 0, 0);        // kraw
  k_scramble<<<4096, 256, 0, stream>>>(sml1, sml2);                  // kT
  gemm_bf16(x_pm, w_v, sml3, bv, NPIX, CRR, CIN, 1, 0, 0, 0);        // v
  transpose_sc<<<4096, 256, 0, stream>>>(sml3, sml1);                // vT

  // 5) attention: logits = q @ k (batched), softmax, out = att @ v
  gemm_f32(sml0, sml2, bigL, nullptr, SS, SS, CRR, BB,
           (long long)SS * CRR, (long long)SS * CRR, (long long)SS * SS);
  softmax_rows<<<BB * SS, 256, 0, stream>>>(bigL, att, SS);
  gemm_bf16(att, sml1, sml2, nullptr, SS, CRR, SS, BB,
            (long long)SS * SS, (long long)CRR * SS, (long long)SS * CRR);

  // 6) value_conv + residual epilogue
  gemm_f32(sml2, w_val, bigL, bval, NPIX, CIN, CRR, 1, 0, 0, 0);     // final (fp32)
  epilogue_k<<<4096, 256, 0, stream>>>(bigL, x, gamma, out, totBig);
}
